// AttnBlock_72292889526543
// MI455X (gfx1250) — matrix operand order repaired and checked by reading, not compiled, not run on hardware
//
#include <hip/hip_runtime.h>
#include <hip/hip_bf16.h>

typedef __attribute__((ext_vector_type(16))) _Float16 v16h;
typedef __attribute__((ext_vector_type(8)))  float    v8f;
typedef __attribute__((ext_vector_type(4)))  unsigned v4u;
typedef __attribute__((ext_vector_type(8)))  int      v8i;
typedef __attribute__((ext_vector_type(4)))  int      v4i;

#define BATCH 4
#define CCH   512
#define NSP   4096
#define NGRP  32

__device__ __forceinline__ v8f wmma_f16(v16h a, v16h b, v8f c) {
    return __builtin_amdgcn_wmma_f32_16x16x32_f16(false, a, false, b,
                                                  (short)0, c, false, false);
}

__device__ __forceinline__ v8f vzero8() {
    v8f z;
#pragma unroll
    for (int i = 0; i < 8; ++i) z[i] = 0.f;
    return z;
}

// A-matrix (16x32 f16) K index for VGPR v, lane-high bit (ISA 7.12.2)
__device__ __forceinline__ int a_kidx(int v, int hi) {
    return ((v & 3) << 1) + ((v >> 2) << 4) + (hi << 3);
}
// B-matrix (32x16 f16) K index
__device__ __forceinline__ int b_kidx(int v, int hi) {
    return ((v & 3) << 1) + ((v >> 2) << 4) + (hi << 3);
}

__device__ __forceinline__ void async_b128(unsigned lds_off, const void* g) {
    unsigned long long ga = (unsigned long long)(uintptr_t)g;
    asm volatile("global_load_async_to_lds_b128 %0, %1, off"
                 :: "v"(lds_off), "v"(ga) : "memory");
}

// ---------------------------------------------------------------------------
// Tensor Data Mover: 2D tile load (global -> LDS), 2-byte elements, with LDS
// row padding done by the TDM (pad_enable).  D# layout per CDNA5 ISA.
//   tile_d0 : elements per row, tile_d1 : rows
//   stride0 : tensor dim0 stride in elements
//   pi/pa   : pad_interval code, pad_amount code
// ---------------------------------------------------------------------------
__device__ __forceinline__ void tdm_load_2d(unsigned lds_addr, const void* gaddr,
                                            unsigned tile_d0, unsigned tile_d1,
                                            unsigned stride0,
                                            unsigned pi, unsigned pa) {
    unsigned long long ga = (unsigned long long)(uintptr_t)gaddr;
    v4u g0;
    g0[0] = 1u;                                          // count=1 (valid user D#)
    g0[1] = lds_addr;                                    // lds_addr
    g0[2] = (unsigned)(ga & 0xffffffffu);                // global_addr[31:0]
    g0[3] = (unsigned)((ga >> 32) & 0x01ffffffu)         // global_addr[56:32]
          | (2u << 30);                                  // type = 2 (image)
    v8i g1;
    g1[0] = (int)((1u << 16)      |                      // data_size = 2 bytes
                  (1u << 20)      |                      // pad_enable
                  (pi << 22)      |                      // pad_interval
                  (pa << 25));                           // pad_amount
    g1[1] = (int)((tile_d0 & 0xffffu) << 16);            // tensor_dim0[15:0]
    g1[2] = (int)(((tile_d0 >> 16) & 0xffffu)            // tensor_dim0[31:16]
                | ((tile_d1 & 0xffffu) << 16));          // tensor_dim1[15:0]
    g1[3] = (int)(((tile_d1 >> 16) & 0xffffu)            // tensor_dim1[31:16]
                | ((tile_d0 & 0xffffu) << 16));          // tile_dim0
    g1[4] = (int)(tile_d1 & 0xffffu);                    // tile_dim1 (tile_dim2=0)
    g1[5] = (int)stride0;                                // tensor_dim0_stride[31:0]
    g1[6] = 0;                                           // stride0 hi | dim1_stride lo
    g1[7] = 0;
    v4i z4 = {0, 0, 0, 0};
#if defined(__clang_major__) && (__clang_major__ >= 23)
    v8i z8 = {0, 0, 0, 0, 0, 0, 0, 0};
    __builtin_amdgcn_tensor_load_to_lds(g0, g1, z4, z4, z8, 0);
#else
    __builtin_amdgcn_tensor_load_to_lds(g0, g1, z4, z4, 0);
#endif
}

union AF { v16h h; unsigned u[8]; };

// ---------------------------------------------------------------------------
// Kernel 1: GroupNorm statistics (mean, rstd) per (batch, group)
// ---------------------------------------------------------------------------
__global__ __launch_bounds__(256)
void gn_stats_kernel(const float* __restrict__ x, float* __restrict__ stats) {
    const int bg = blockIdx.x;                 // b*NGRP + g
    const float* p = x + (size_t)bg * 16 * NSP;
    float s = 0.f, ss = 0.f;
    for (int i = threadIdx.x; i < 16 * NSP; i += 256) {
        float v = p[i];
        s += v; ss += v * v;
    }
    __shared__ float rs[256], rss[256];
    rs[threadIdx.x] = s; rss[threadIdx.x] = ss;
    __syncthreads();
    for (int o = 128; o > 0; o >>= 1) {
        if (threadIdx.x < o) {
            rs[threadIdx.x]  += rs[threadIdx.x + o];
            rss[threadIdx.x] += rss[threadIdx.x + o];
        }
        __syncthreads();
    }
    if (threadIdx.x == 0) {
        const float inv = 1.f / (16.f * NSP);
        float mean = rs[0] * inv;
        float var  = rss[0] * inv - mean * mean;
        stats[bg * 2 + 0] = mean;
        stats[bg * 2 + 1] = rsqrtf(var + 1e-6f);
    }
}

// ---------------------------------------------------------------------------
// Kernel 2: fused GroupNorm-apply + 1x1 conv GEMM (W[o,c] x Xn[c,p])
//   TRANS=1: store out as (p, o) f16  (used for Q, K; scale folds c^-0.5)
//   TRANS=0: store out as (o, p) f16  (used for V)
// ---------------------------------------------------------------------------
template <bool TRANS>
__global__ __launch_bounds__(256)
void qkv_kernel(const float* __restrict__ x, const float* __restrict__ stats,
                const float* __restrict__ gamma, const float* __restrict__ beta,
                const float* __restrict__ W, const float* __restrict__ bias,
                _Float16* __restrict__ out, float scale) {
    const int nBase = blockIdx.x * 128;
    const int mBase = blockIdx.y * 128;
    const int b     = blockIdx.z;
    const float*    xb   = x   + (size_t)b * CCH * NSP;
    _Float16*       outb = out + (size_t)b * CCH * NSP;
    const int t = threadIdx.x, lane = t & 31, wave = t >> 5;
    const int hi = (lane >> 4) & 1, l16 = lane & 15;
    const int wm = (wave >> 2) * 64;   // 0 or 64
    const int wn = (wave & 3) * 32;    // 0,32,64,96

    __shared__ _Float16 Wt[128][36];   // A tile [o][k]
    __shared__ _Float16 Xt[128][36];   // B tile [p][k] (K-major!)
    __shared__ float    Bs[128];

    if (t < 128) Bs[t] = bias[mBase + t];

    v8f acc[4][2];
#pragma unroll
    for (int mi = 0; mi < 4; ++mi)
#pragma unroll
        for (int ni = 0; ni < 2; ++ni) acc[mi][ni] = vzero8();

    for (int k0 = 0; k0 < CCH; k0 += 32) {
        __syncthreads();
#pragma unroll
        for (int it = 0; it < 16; ++it) {       // W slab 128x32
            int idx = t + it * 256;
            int row = idx >> 5, col = idx & 31;
            Wt[row][col] = (_Float16)W[(size_t)(mBase + row) * CCH + k0 + col];
        }
#pragma unroll
        for (int it = 0; it < 16; ++it) {       // X slab 32x128, normalized
            int idx = t + it * 256;
            int kk = idx >> 7, p = idx & 127;
            int cg = k0 + kk;
            float mean = stats[(b * NGRP + (cg >> 4)) * 2 + 0];
            float rstd = stats[(b * NGRP + (cg >> 4)) * 2 + 1];
            float v = xb[(size_t)cg * NSP + nBase + p];
            v = (v - mean) * rstd * gamma[cg] + beta[cg];
            Xt[p][kk] = (_Float16)v;
        }
        __syncthreads();

        AF a[4], bf[2];
#pragma unroll
        for (int mi = 0; mi < 4; ++mi) {
            int row = wm + mi * 16 + l16;
#pragma unroll
            for (int v = 0; v < 8; ++v)
                a[mi].u[v] = *(const unsigned*)&Wt[row][a_kidx(v, hi)];
        }
#pragma unroll
        for (int ni = 0; ni < 2; ++ni) {
            int p = wn + ni * 16 + l16;
#pragma unroll
            for (int v = 0; v < 8; ++v)
                bf[ni].u[v] = *(const unsigned*)&Xt[p][b_kidx(v, hi)];
        }
#pragma unroll
        for (int mi = 0; mi < 4; ++mi)
#pragma unroll
            for (int ni = 0; ni < 2; ++ni)
                acc[mi][ni] = wmma_f16(a[mi].h, bf[ni].h, acc[mi][ni]);
    }

#pragma unroll
    for (int mi = 0; mi < 4; ++mi)
#pragma unroll
        for (int ni = 0; ni < 2; ++ni) {
            if (TRANS) {
                int p  = nBase + wn + ni * 16 + l16;
                int o0 = wm + mi * 16 + hi * 8;
                alignas(16) _Float16 h8[8];
#pragma unroll
                for (int r = 0; r < 8; ++r)
                    h8[r] = (_Float16)((acc[mi][ni][r] + Bs[o0 + r]) * scale);
                *(float4*)&outb[(size_t)p * CCH + mBase + o0] = *(float4*)h8;
            } else {
#pragma unroll
                for (int r = 0; r < 8; ++r) {
                    int o = wm + mi * 16 + hi * 8 + r;
                    int p = wn + ni * 16 + l16;
                    outb[(size_t)(mBase + o) * NSP + nBase + p] =
                        (_Float16)((acc[mi][ni][r] + Bs[o]) * scale);
                }
            }
        }
}

// ---------------------------------------------------------------------------
// Kernel 3: flash attention.  Block = 64 queries, full d=512.
//   Qt,Kt are (n,c) f16; Vc is (c,n) f16; Oc written (c,n) f16.
//   8 waves; wave w owns channel slice [64w, 64w+64).
//   K/V tiles staged by the Tensor Data Mover (one descriptor per tile, LDS
//   row padding done in hardware); Q tile staged with async global->LDS.
// ---------------------------------------------------------------------------
__global__ __launch_bounds__(256)
void flash_kernel(const _Float16* __restrict__ Qt, const _Float16* __restrict__ Kt,
                  const _Float16* __restrict__ Vc, _Float16* __restrict__ Oc) {
    const int qBase = blockIdx.x * 64;
    const int b     = blockIdx.y;
    const _Float16* Qb = Qt + (size_t)b * CCH * NSP;
    const _Float16* Kb = Kt + (size_t)b * CCH * NSP;
    const _Float16* Vb = Vc + (size_t)b * CCH * NSP;
    _Float16*       Ob = Oc + (size_t)b * CCH * NSP;
    const int t = threadIdx.x, lane = t & 31, wave = t >> 5;
    const int hi = (lane >> 4) & 1, l16 = lane & 15;
    const int chw = wave * 64;

    __shared__ _Float16 Qs[64][520];   // [query][ch]   rows 1040B (1024 + 16 pad)
    __shared__ _Float16 Ks[32][520];   // [key][ch]     rows 1040B (1024 + 16 pad)
    __shared__ _Float16 Vs[512][40];   // [ch][j]       rows   80B (  64 + 16 pad)
    __shared__ float    Sp[64][32];    // score tile (f32, cross-wave reduced)
    __shared__ _Float16 Pf[64][40];    // softmax probs f16
    __shared__ float    rowM[64], rowL[64], rowSc[64];

    const unsigned qs0 = (unsigned)(uintptr_t)&Qs[0][0];
    const unsigned ks0 = (unsigned)(uintptr_t)&Ks[0][0];
    const unsigned vs0 = (unsigned)(uintptr_t)&Vs[0][0];

    // async-load Q tile: 64 rows x 1KB
#pragma unroll
    for (int it = 0; it < 16; ++it) {
        int cid = t + it * 256;
        int row = cid >> 6, seg = cid & 63;
        async_b128(qs0 + (unsigned)(row * 520 + seg * 8) * 2,
                   Qb + (size_t)(qBase + row) * CCH + seg * 8);
    }
    if (t < 64) { rowM[t] = -1e30f; rowL[t] = 0.f; }

    v8f O[4][4];
#pragma unroll
    for (int mi = 0; mi < 4; ++mi)
#pragma unroll
        for (int ni = 0; ni < 4; ++ni) O[mi][ni] = vzero8();

    asm volatile("s_wait_asynccnt 0x0" ::: "memory");
    __syncthreads();

    for (int kt = 0; kt < NSP / 32; ++kt) {
        const int kBase = kt * 32;
        // TDM: K tile = 32 rows x 512 ch (row 1024B, +16B LDS pad: pi=7, pa=3)
        //      V tile = 512 rows x 32 j  (row   64B, +16B LDS pad: pi=3, pa=3)
        if (wave == 0) {
            tdm_load_2d(ks0, Kb + (size_t)kBase * CCH, 512, 32, 512, 7, 3);
            tdm_load_2d(vs0, Vb + kBase, 32, 512, NSP, 3, 3);
        }
        // zero score tile while TDM is in flight
#pragma unroll
        for (int it = 0; it < 8; ++it) ((float*)Sp)[t + it * 256] = 0.f;

        if (wave == 0) __builtin_amdgcn_s_wait_tensorcnt((short)1);  // K done
        __syncthreads();

        // partial scores over this wave's 64-channel slice
        v8f S[4][2];
#pragma unroll
        for (int mi = 0; mi < 4; ++mi)
#pragma unroll
            for (int nj = 0; nj < 2; ++nj) S[mi][nj] = vzero8();
#pragma unroll
        for (int ks = 0; ks < 2; ++ks) {
            const int cb = chw + ks * 32;
            AF a[4], bf[2];
#pragma unroll
            for (int mi = 0; mi < 4; ++mi) {
                int row = mi * 16 + l16;
#pragma unroll
                for (int v = 0; v < 8; ++v)
                    a[mi].u[v] = *(const unsigned*)&Qs[row][cb + a_kidx(v, hi)];
            }
#pragma unroll
            for (int nj = 0; nj < 2; ++nj) {
                int col = nj * 16 + l16;
#pragma unroll
                for (int v = 0; v < 8; ++v)
                    bf[nj].u[v] = *(const unsigned*)&Ks[col][cb + b_kidx(v, hi)];
            }
#pragma unroll
            for (int mi = 0; mi < 4; ++mi)
#pragma unroll
                for (int nj = 0; nj < 2; ++nj)
                    S[mi][nj] = wmma_f16(a[mi].h, bf[nj].h, S[mi][nj]);
        }
        // cross-wave reduce via LDS float atomics (ds_add_f32)
#pragma unroll
        for (int mi = 0; mi < 4; ++mi)
#pragma unroll
            for (int nj = 0; nj < 2; ++nj)
#pragma unroll
                for (int r = 0; r < 8; ++r)
                    atomicAdd(&Sp[mi * 16 + hi * 8 + r][nj * 16 + l16],
                              S[mi][nj][r]);
        __syncthreads();

        // online softmax (scale already folded into Q)
        if (t < 64) {
            float m0 = rowM[t], m = m0;
#pragma unroll 8
            for (int j = 0; j < 32; ++j) m = fmaxf(m, Sp[t][j]);
            float sc = __expf(m0 - m);
            float lsum = 0.f;
#pragma unroll 8
            for (int j = 0; j < 32; ++j) {
                float p = __expf(Sp[t][j] - m);
                Pf[t][j] = (_Float16)p;
                lsum += p;
            }
            rowM[t] = m;
            rowL[t] = rowL[t] * sc + lsum;
            rowSc[t] = sc;
        }
        if (wave == 0) __builtin_amdgcn_s_wait_tensorcnt((short)0);  // V done
        __syncthreads();

        // rescale accumulators by exp(m_old - m_new)
        float scv[4][8];
#pragma unroll
        for (int mi = 0; mi < 4; ++mi)
#pragma unroll
            for (int r = 0; r < 8; ++r)
                scv[mi][r] = rowSc[mi * 16 + hi * 8 + r];
#pragma unroll
        for (int mi = 0; mi < 4; ++mi)
#pragma unroll
            for (int ni = 0; ni < 4; ++ni)
#pragma unroll
                for (int r = 0; r < 8; ++r) O[mi][ni][r] *= scv[mi][r];

        // O += P x V  (K-dim = 32 keys)
        AF ap[4], bv[4];
#pragma unroll
        for (int mi = 0; mi < 4; ++mi) {
            int row = mi * 16 + l16;
#pragma unroll
            for (int v = 0; v < 8; ++v)
                ap[mi].u[v] = *(const unsigned*)&Pf[row][a_kidx(v, hi)];
        }
#pragma unroll
        for (int ni = 0; ni < 4; ++ni) {
            int ch = chw + ni * 16 + l16;
#pragma unroll
            for (int v = 0; v < 8; ++v)
                bv[ni].u[v] = *(const unsigned*)&Vs[ch][b_kidx(v, hi)];
        }
#pragma unroll
        for (int mi = 0; mi < 4; ++mi)
#pragma unroll
            for (int ni = 0; ni < 4; ++ni)
                O[mi][ni] = wmma_f16(ap[mi].h, bv[ni].h, O[mi][ni]);

        __syncthreads();   // protect LDS tiles before next iteration's DMA
    }

    // finalize: divide by l, store O as (c, n) f16 (contiguous 16B per tile)
    float inv[4][8];
#pragma unroll
    for (int mi = 0; mi < 4; ++mi)
#pragma unroll
        for (int r = 0; r < 8; ++r)
            inv[mi][r] = 1.f / rowL[mi * 16 + hi * 8 + r];
#pragma unroll
    for (int mi = 0; mi < 4; ++mi)
#pragma unroll
        for (int ni = 0; ni < 4; ++ni) {
            alignas(16) _Float16 h8[8];
#pragma unroll
            for (int r = 0; r < 8; ++r)
                h8[r] = (_Float16)(O[mi][ni][r] * inv[mi][r]);
            int ch = chw + ni * 16 + l16;
            *(float4*)&Ob[(size_t)ch * NSP + qBase + mi * 16 + hi * 8] =
                *(float4*)h8;
        }
}

// ---------------------------------------------------------------------------
// Kernel 4: projection GEMM + bias + residual, f32 output
// ---------------------------------------------------------------------------
__global__ __launch_bounds__(256)
void proj_kernel(const _Float16* __restrict__ Oc, const float* __restrict__ W,
                 const float* __restrict__ bias, const float* __restrict__ x,
                 float* __restrict__ out) {
    const int nBase = blockIdx.x * 128;
    const int mBase = blockIdx.y * 128;
    const int b     = blockIdx.z;
    const _Float16* Ob = Oc + (size_t)b * CCH * NSP;
    const float*    xb = x  + (size_t)b * CCH * NSP;
    float*          ob = out + (size_t)b * CCH * NSP;
    const int t = threadIdx.x, lane = t & 31, wave = t >> 5;
    const int hi = (lane >> 4) & 1, l16 = lane & 15;
    const int wm = (wave >> 2) * 64;
    const int wn = (wave & 3) * 32;

    __shared__ _Float16 Wt[128][36];
    __shared__ _Float16 Xt[128][36];
    __shared__ float    Bs[128];

    if (t < 128) Bs[t] = bias[mBase + t];

    v8f acc[4][2];
#pragma unroll
    for (int mi = 0; mi < 4; ++mi)
#pragma unroll
        for (int ni = 0; ni < 2; ++ni) acc[mi][ni] = vzero8();

    for (int k0 = 0; k0 < CCH; k0 += 32) {
        __syncthreads();
#pragma unroll
        for (int it = 0; it < 16; ++it) {
            int idx = t + it * 256;
            int row = idx >> 5, col = idx & 31;
            Wt[row][col] = (_Float16)W[(size_t)(mBase + row) * CCH + k0 + col];
        }
#pragma unroll
        for (int it = 0; it < 16; ++it) {
            int idx = t + it * 256;
            int kk = idx >> 7, p = idx & 127;
            Xt[p][kk] = Ob[(size_t)(k0 + kk) * NSP + nBase + p];
        }
        __syncthreads();

        AF a[4], bf[2];
#pragma unroll
        for (int mi = 0; mi < 4; ++mi) {
            int row = wm + mi * 16 + l16;
#pragma unroll
            for (int v = 0; v < 8; ++v)
                a[mi].u[v] = *(const unsigned*)&Wt[row][a_kidx(v, hi)];
        }
#pragma unroll
        for (int ni = 0; ni < 2; ++ni) {
            int p = wn + ni * 16 + l16;
#pragma unroll
            for (int v = 0; v < 8; ++v)
                bf[ni].u[v] = *(const unsigned*)&Xt[p][b_kidx(v, hi)];
        }
#pragma unroll
        for (int mi = 0; mi < 4; ++mi)
#pragma unroll
            for (int ni = 0; ni < 2; ++ni)
                acc[mi][ni] = wmma_f16(a[mi].h, bf[ni].h, acc[mi][ni]);
    }

#pragma unroll
    for (int mi = 0; mi < 4; ++mi)
#pragma unroll
        for (int ni = 0; ni < 2; ++ni)
#pragma unroll
            for (int r = 0; r < 8; ++r) {
                int o = wm + mi * 16 + hi * 8 + r;
                int p = wn + ni * 16 + l16;
                size_t idx = (size_t)(mBase + o) * NSP + nBase + p;
                ob[idx] = xb[idx] + acc[mi][ni][r] + Bs[o];
            }
}

// ---------------------------------------------------------------------------
extern "C" void kernel_launch(void* const* d_in, const int* in_sizes, int n_in,
                              void* d_out, int out_size, void* d_ws, size_t ws_size,
                              hipStream_t stream) {
    const float* x     = (const float*)d_in[0];
    const float* gamma = (const float*)d_in[1];
    const float* beta  = (const float*)d_in[2];
    const float* Wq    = (const float*)d_in[3];
    const float* bq    = (const float*)d_in[4];
    const float* Wk    = (const float*)d_in[5];
    const float* bk    = (const float*)d_in[6];
    const float* Wv    = (const float*)d_in[7];
    const float* bv    = (const float*)d_in[8];
    const float* Wp    = (const float*)d_in[9];
    const float* bp    = (const float*)d_in[10];
    float* out = (float*)d_out;

    char* ws = (char*)d_ws;
    const size_t matBytes = (size_t)BATCH * CCH * NSP * sizeof(_Float16); // 16 MB
    float*     stats = (float*)ws;
    _Float16*  Qt    = (_Float16*)(ws + 4096);
    _Float16*  Kt    = (_Float16*)(ws + 4096 + matBytes);
    _Float16*  Vc    = (_Float16*)(ws + 4096 + 2 * matBytes);
    _Float16*  Oc    = (_Float16*)(ws + 4096 + 3 * matBytes);

    gn_stats_kernel<<<BATCH * NGRP, 256, 0, stream>>>(x, stats);

    const dim3 g(NSP / 128, CCH / 128, BATCH);
    const float qscale = 0.044194173824159216f;  // 512^-0.5
    qkv_kernel<true ><<<g, 256, 0, stream>>>(x, stats, gamma, beta, Wq, bq, Qt, qscale);
    qkv_kernel<true ><<<g, 256, 0, stream>>>(x, stats, gamma, beta, Wk, bk, Kt, 1.f);
    qkv_kernel<false><<<g, 256, 0, stream>>>(x, stats, gamma, beta, Wv, bv, Vc, 1.f);

    flash_kernel<<<dim3(NSP / 64, BATCH), 256, 0, stream>>>(Qt, Kt, Vc, Oc);

    proj_kernel<<<g, 256, 0, stream>>>(Oc, Wp, bp, x, out);
}
